// Attention_6992206758268
// MI455X (gfx1250) — matrix operand order repaired and checked by reading, not compiled, not run on hardware
//
#include <hip/hip_runtime.h>
#include <stdint.h>

// Problem constants (match reference)
#define DIM   768
#define NHEAD 12
#define HDIM  64
#define BATCH 2
#define SEQ   2048
#define MROWS (BATCH * SEQ)   // 4096 rows of x flattened
#define C3    (3 * DIM)       // 2304

typedef __attribute__((ext_vector_type(16))) _Float16 v16h;
typedef __attribute__((ext_vector_type(8)))  _Float16 v8h;
typedef __attribute__((ext_vector_type(8)))  float    v8f;
typedef __attribute__((ext_vector_type(4)))  int      v4i;

// Concatenate two v8h into the v16h operand register block.
static __device__ __forceinline__ v16h cat8(v8h lo, v8h hi) {
  return __builtin_shufflevector(lo, hi, 0,1,2,3,4,5,6,7,8,9,10,11,12,13,14,15);
}

static __device__ __forceinline__ v8f wmma_f16(v16h a, v16h b, v8f c) {
  // D = A(16x32 f16) * B(32x16 f16) + C(16x16 f32)
  return __builtin_amdgcn_wmma_f32_16x16x32_f16(false, a, false, b, (short)0, c,
                                                false, false);
}

// Async global->LDS 16-byte copy (ASYNCcnt-tracked).
// Builtin signature on this toolchain: (v4i global*, v4i local*, int off, int cpol).
static __device__ __forceinline__ void async_b128(_Float16* lds, const _Float16* g) {
#if __has_builtin(__builtin_amdgcn_global_load_async_to_lds_b128)
  __builtin_amdgcn_global_load_async_to_lds_b128(
      (__attribute__((address_space(1))) v4i*)g,
      (__attribute__((address_space(3))) v4i*)lds, 0, 0);
#else
  unsigned loff = (unsigned)(uintptr_t)(__attribute__((address_space(3))) void*)lds;
  asm volatile("global_load_async_to_lds_b128 %0, %1, off"
               :: "v"(loff), "v"(g) : "memory");
#endif
}

static __device__ __forceinline__ void wait_async0() {
#if __has_builtin(__builtin_amdgcn_s_wait_asynccnt)
  __builtin_amdgcn_s_wait_asynccnt(0);
#else
  asm volatile("s_wait_asynccnt 0" ::: "memory");
#endif
}

// Row-wise reductions across the 16 lanes of a half-wave (wave32).
static __device__ __forceinline__ float red_max16(float v) {
#pragma unroll
  for (int m = 1; m < 16; m <<= 1) v = fmaxf(v, __shfl_xor(v, m, 32));
  return v;
}
static __device__ __forceinline__ float red_sum16(float v) {
#pragma unroll
  for (int m = 1; m < 16; m <<= 1) v += __shfl_xor(v, m, 32);
  return v;
}

// ---------------- conversion kernels ----------------
__global__ void k_cvt(const float* __restrict__ in, _Float16* __restrict__ out, int n) {
  int i = blockIdx.x * blockDim.x + threadIdx.x;
  if (i < n) out[i] = (_Float16)in[i];
}

// in: [R][C] f32 row-major  ->  out: [C][R] f16 (transposed)
__global__ void k_cvt_tr(const float* __restrict__ in, _Float16* __restrict__ out,
                         int R, int C) {
  int i = blockIdx.x * blockDim.x + threadIdx.x;
  if (i < R * C) {
    int r = i / C, c = i - r * C;
    out[(size_t)c * R + r] = (_Float16)in[i];
  }
}

// ---------------- QKV projection GEMM ----------------
// C[4096 x 2304] = XH[4096 x 768] * W  (WT given transposed: [2304][768])
// 4 waves / block; each wave owns a 32(M) x 64(N) strip (2x M register blocking
// so every B fragment feeds two WMMAs). Epilogue: +bias, scatter to Q/K/V^T.
// The 64-col panel is block-uniform in (s, h); the 128-row panel is batch-uniform.
__global__ __launch_bounds__(128) void k_qkv(
    const _Float16* __restrict__ XH, const _Float16* __restrict__ WT,
    const float* __restrict__ bias,
    _Float16* __restrict__ Qh, _Float16* __restrict__ Kh, _Float16* __restrict__ Vt) {
  const int wave = threadIdx.x >> 5;
  const int lane = threadIdx.x & 31;
  const int hw = lane >> 4, l16 = lane & 15;
  const int m0 = blockIdx.x * 128 + wave * 32;
  const int n0 = blockIdx.y * 64;

  v8f acc0[4] = {}, acc1[4] = {};
  const _Float16* arow0 = XH + (size_t)(m0 + l16) * DIM + 16 * hw;
  const _Float16* arow1 = arow0 + (size_t)16 * DIM;

  for (int k0 = 0; k0 < DIM; k0 += 32) {
    v16h A0 = cat8(*(const v8h*)(arow0 + k0), *(const v8h*)(arow0 + k0 + 8));
    v16h A1 = cat8(*(const v8h*)(arow1 + k0), *(const v8h*)(arow1 + k0 + 8));
    __builtin_prefetch(arow0 + k0 + 64, 0, 3);
#pragma unroll
    for (int t = 0; t < 4; ++t) {
      const _Float16* bp = WT + (size_t)(n0 + t * 16 + l16) * DIM + k0 + 16 * hw;
      v16h Bf = *(const v16h*)bp;
      acc0[t] = wmma_f16(A0, Bf, acc0[t]);
      acc1[t] = wmma_f16(A1, Bf, acc1[t]);
    }
  }

  // Block-uniform decomposition: s = Q/K/V selector, h = head.
  const int s = blockIdx.y / NHEAD;           // 0=Q, 1=K, 2=V
  const int h = blockIdx.y - s * NHEAD;       // [0, 12)
  const int bidx = m0 >> 11;                  // batch (128 | 2048 so uniform)
  const int nbase = m0 & (SEQ - 1);
  const size_t bh = (size_t)(bidx * NHEAD + h);

  if (s == 0) {
    _Float16* dst = Qh + (bh * SEQ + nbase) * HDIM;
#pragma unroll
    for (int rt = 0; rt < 2; ++rt)
#pragma unroll
      for (int t = 0; t < 4; ++t) {
        int d = t * 16 + l16;
        float bv = bias[n0 + d];
#pragma unroll
        for (int r = 0; r < 8; ++r) {
          int n = rt * 16 + r + 8 * hw;
          float val = (rt ? acc1[t][r] : acc0[t][r]) + bv;
          dst[(size_t)n * HDIM + d] = (_Float16)(val * 0.125f);
        }
      }
  } else if (s == 1) {
    _Float16* dst = Kh + (bh * SEQ + nbase) * HDIM;
#pragma unroll
    for (int rt = 0; rt < 2; ++rt)
#pragma unroll
      for (int t = 0; t < 4; ++t) {
        int d = t * 16 + l16;
        float bv = bias[n0 + d];
#pragma unroll
        for (int r = 0; r < 8; ++r) {
          int n = rt * 16 + r + 8 * hw;
          dst[(size_t)n * HDIM + d] = (_Float16)((rt ? acc1[t][r] : acc0[t][r]) + bv);
        }
      }
  } else {
    _Float16* dst = Vt + bh * HDIM * SEQ + nbase;   // [64][SEQ], transposed
#pragma unroll
    for (int rt = 0; rt < 2; ++rt)
#pragma unroll
      for (int t = 0; t < 4; ++t) {
        int d = t * 16 + l16;
        float bv = bias[n0 + d];
#pragma unroll
        for (int r = 0; r < 8; ++r) {
          int n = rt * 16 + r + 8 * hw;
          dst[(size_t)d * SEQ + n] = (_Float16)((rt ? acc1[t][r] : acc0[t][r]) + bv);
        }
      }
  }
}

// ---------------- flash attention ----------------
// Per (b, h): Q,K row-major [SEQ][64] f16 (Q pre-scaled), V transposed [64][SEQ].
// Block = 4 waves; each wave owns a 16-row query tile. Per 32-key block the
// K (32x64) and V^T (64x32) tiles are staged once per block into LDS with
// async global->LDS copies (shared by all 4 waves: 4x traffic cut).
__global__ __launch_bounds__(128) void k_attn(
    const _Float16* __restrict__ Qh, const _Float16* __restrict__ Kh,
    const _Float16* __restrict__ Vt, _Float16* __restrict__ Oh) {
  __shared__ __align__(32) _Float16 kbuf[32 * HDIM];    // rows j-local, 64 dims
  __shared__ __align__(32) _Float16 vbuf[HDIM * 32];    // rows d, 32 j-cols
  __shared__ __align__(32) _Float16 pbuf[4][16 * 32];   // per-wave P relayout
  const int tid = threadIdx.x;
  const int wave = tid >> 5;
  const int lane = tid & 31;
  const int hw = lane >> 4, l16 = lane & 15;
  const int b = blockIdx.z, h = blockIdx.y;
  const int q0 = blockIdx.x * 64 + wave * 16;
  const size_t bh = (size_t)(b * NHEAD + h);
  const _Float16* Q = Qh + bh * SEQ * HDIM;
  const _Float16* K = Kh + bh * SEQ * HDIM;
  const _Float16* V = Vt + bh * HDIM * SEQ;             // [64][SEQ]
  _Float16* pb = pbuf[wave];

  // Resident Q A-fragments (d-chunks 0..31, 32..63)
  v16h qa[2];
#pragma unroll
  for (int c = 0; c < 2; ++c) {
    const _Float16* qp = Q + (size_t)(q0 + l16) * HDIM + c * 32 + 16 * hw;
    qa[c] = cat8(*(const v8h*)qp, *(const v8h*)(qp + 8));
  }

  v8f o0 = {}, o1 = {}, o2 = {}, o3 = {};
  float m[8], l[8];
#pragma unroll
  for (int r = 0; r < 8; ++r) { m[r] = -3.0e38f; l[r] = 0.f; }

  for (int j0 = 0; j0 < SEQ; j0 += 32) {
    // Cooperative async stage: K tile is 4 KB contiguous; V tile is 64 x 64B rows.
    const _Float16* kg = K + (size_t)j0 * HDIM;
#pragma unroll
    for (int c = tid; c < 256; c += 128) {
      async_b128(&kbuf[c * 8], kg + c * 8);
      int d = c >> 2, sub = c & 3;
      async_b128(&vbuf[d * 32 + sub * 8], V + (size_t)d * SEQ + j0 + sub * 8);
    }
    wait_async0();       // my async copies landed
    __syncthreads();     // everyone's landed

    // S = Q * K^T for 32 keys: two 16x16 score tiles, 2 k-chunks each.
    v8f s1 = {}, s2 = {};
#pragma unroll
    for (int c = 0; c < 2; ++c) {
      v16h b1 = *(const v16h*)&kbuf[l16 * HDIM + c * 32 + 16 * hw];
      v16h b2 = *(const v16h*)&kbuf[(16 + l16) * HDIM + c * 32 + 16 * hw];
      s1 = wmma_f16(qa[c], b1, s1);
      s2 = wmma_f16(qa[c], b2, s2);
    }

    // Online softmax (rows live in accumulator elements; cols across 16 lanes).
#pragma unroll
    for (int r = 0; r < 8; ++r) {
      float smax  = red_max16(fmaxf(s1[r], s2[r]));
      float mn    = fmaxf(m[r], smax);
      float alpha = __expf(m[r] - mn);
      float p1    = __expf(s1[r] - mn);
      float p2    = __expf(s2[r] - mn);
      float rs    = red_sum16(p1 + p2);
      l[r] = l[r] * alpha + rs;
      m[r] = mn;
      o0[r] *= alpha; o1[r] *= alpha; o2[r] *= alpha; o3[r] *= alpha;
      // D-layout -> row-major 16x32 P tile in LDS (per-wave region).
      pb[(r + 8 * hw) * 32 + l16]      = (_Float16)p1;
      pb[(r + 8 * hw) * 32 + 16 + l16] = (_Float16)p2;
    }
    asm volatile("s_wait_dscnt 0" ::: "memory");  // same-wave P stores visible

    // Reload P as an A-fragment, then O += P(16x32) * V(32x64).
    const _Float16* pr = pb + l16 * 32 + 16 * hw;
    v16h Pa = cat8(*(const v8h*)pr, *(const v8h*)(pr + 8));
    o0 = wmma_f16(Pa, *(const v16h*)&vbuf[(0 * 16 + l16) * 32 + 16 * hw], o0);
    o1 = wmma_f16(Pa, *(const v16h*)&vbuf[(1 * 16 + l16) * 32 + 16 * hw], o1);
    o2 = wmma_f16(Pa, *(const v16h*)&vbuf[(2 * 16 + l16) * 32 + 16 * hw], o2);
    o3 = wmma_f16(Pa, *(const v16h*)&vbuf[(3 * 16 + l16) * 32 + 16 * hw], o3);

    asm volatile("s_wait_dscnt 0" ::: "memory");  // tile reads done
    __syncthreads();                              // before next overwrite
  }

  // Normalize and store merged-head row-major f16 [B*N][DIM].
#pragma unroll
  for (int r = 0; r < 8; ++r) {
    float inv = 1.0f / l[r];
    size_t row = (size_t)b * SEQ + q0 + r + 8 * hw;
    _Float16* op = Oh + row * DIM + h * HDIM + l16;
    op[0]  = (_Float16)(o0[r] * inv);
    op[16] = (_Float16)(o1[r] * inv);
    op[32] = (_Float16)(o2[r] * inv);
    op[48] = (_Float16)(o3[r] * inv);
  }
}

// ---------------- output projection GEMM ----------------
// out[4096 x 768] f32 = AH[4096 x 768] * Wp (WT transposed [768][768]) + bias
__global__ __launch_bounds__(128) void k_proj(
    const _Float16* __restrict__ AH, const _Float16* __restrict__ WT,
    const float* __restrict__ bias, float* __restrict__ out) {
  const int wave = threadIdx.x >> 5;
  const int lane = threadIdx.x & 31;
  const int hw = lane >> 4, l16 = lane & 15;
  const int m0 = blockIdx.x * 128 + wave * 32;
  const int n0 = blockIdx.y * 64;

  v8f acc0[4] = {}, acc1[4] = {};
  const _Float16* arow0 = AH + (size_t)(m0 + l16) * DIM + 16 * hw;
  const _Float16* arow1 = arow0 + (size_t)16 * DIM;

  for (int k0 = 0; k0 < DIM; k0 += 32) {
    v16h A0 = cat8(*(const v8h*)(arow0 + k0), *(const v8h*)(arow0 + k0 + 8));
    v16h A1 = cat8(*(const v8h*)(arow1 + k0), *(const v8h*)(arow1 + k0 + 8));
    __builtin_prefetch(arow0 + k0 + 64, 0, 3);
#pragma unroll
    for (int t = 0; t < 4; ++t) {
      const _Float16* bp = WT + (size_t)(n0 + t * 16 + l16) * DIM + k0 + 16 * hw;
      v16h Bf = *(const v16h*)bp;
      acc0[t] = wmma_f16(A0, Bf, acc0[t]);
      acc1[t] = wmma_f16(A1, Bf, acc1[t]);
    }
  }
#pragma unroll
  for (int rt = 0; rt < 2; ++rt) {
#pragma unroll
    for (int t = 0; t < 4; ++t) {
      int col = n0 + t * 16 + l16;
      float bv = bias[col];
#pragma unroll
      for (int r = 0; r < 8; ++r) {
        int row = m0 + rt * 16 + r + 8 * hw;
        out[(size_t)row * DIM + col] = (rt ? acc1[t][r] : acc0[t][r]) + bv;
      }
    }
  }
}

// ---------------- launcher ----------------
extern "C" void kernel_launch(void* const* d_in, const int* in_sizes, int n_in,
                              void* d_out, int out_size, void* d_ws, size_t ws_size,
                              hipStream_t stream) {
  (void)in_sizes; (void)n_in; (void)out_size; (void)ws_size;
  const float* x      = (const float*)d_in[0];
  const float* qkv_w  = (const float*)d_in[1];
  const float* qkv_b  = (const float*)d_in[2];
  const float* proj_w = (const float*)d_in[3];
  const float* proj_b = (const float*)d_in[4];
  float* out = (float*)d_out;

  char* ws = (char*)d_ws;
  size_t off = 0;
  auto take = [&](size_t elems) -> _Float16* {
    void* p = ws + off;
    off = (off + elems * 2 + 255) & ~(size_t)255;
    return (_Float16*)p;
  };
  const size_t perhead = (size_t)BATCH * NHEAD * SEQ * HDIM;
  _Float16* XH  = take((size_t)MROWS * DIM);   // x in f16
  _Float16* WqT = take((size_t)C3 * DIM);      // qkv_w^T f16
  _Float16* WpT = take((size_t)DIM * DIM);     // proj_w^T f16
  _Float16* Qh  = take(perhead);               // [B][H][N][64], pre-scaled
  _Float16* Kh  = take(perhead);               // [B][H][N][64]
  _Float16* Vt  = take(perhead);               // [B][H][64][N] (transposed)
  _Float16* Oh  = take((size_t)MROWS * DIM);   // merged-head attn out f16

  int n = MROWS * DIM;
  k_cvt<<<(n + 255) / 256, 256, 0, stream>>>(x, XH, n);
  n = DIM * C3;
  k_cvt_tr<<<(n + 255) / 256, 256, 0, stream>>>(qkv_w, WqT, DIM, C3);
  n = DIM * DIM;
  k_cvt_tr<<<(n + 255) / 256, 256, 0, stream>>>(proj_w, WpT, DIM, DIM);

  k_qkv <<<dim3(MROWS / 128, C3 / 64),    128, 0, stream>>>(XH, WqT, qkv_b, Qh, Kh, Vt);
  k_attn<<<dim3(SEQ / 64, NHEAD, BATCH),  128, 0, stream>>>(Qh, Kh, Vt, Oh);
  k_proj<<<dim3(MROWS / 128, DIM / 64),   128, 0, stream>>>(Oh, WpT, proj_b, out);
}
